// AcousticGuidedMambaBlock_55568286875769
// MI455X (gfx1250) — matrix operand order repaired and checked by reading, not compiled, not run on hardware
//
#include <hip/hip_runtime.h>
#include <hip/hip_bf16.h>
#include <math.h>

// ---------------- problem constants ----------------
#define BATCH   4
#define SEQ     1024
#define DIM     512
#define D_INNER 1024
#define D_STATE 16
#define DT_RANK 32
#define D_CONV  4
#define MTOK    (BATCH*SEQ)        // 4096 token rows

typedef __bf16 bf16_t;
typedef __attribute__((ext_vector_type(8)))  __bf16 v8bf;
typedef __attribute__((ext_vector_type(16))) __bf16 v16bf;
typedef __attribute__((ext_vector_type(8)))  float  v8f;

#define SHUF16(lo, hi) __builtin_shufflevector(lo, hi, 0,1,2,3,4,5,6,7,8,9,10,11,12,13,14,15)
#define WMMA_BF16(a, b, c) __builtin_amdgcn_wmma_f32_16x16x32_bf16(false, a, false, b, (short)0, c, false, false)

// ---------------- f32 -> bf16 cast ----------------
__global__ void f2bf_kernel(const float* __restrict__ in, bf16_t* __restrict__ out, int n) {
    int i = blockIdx.x * 256 + threadIdx.x;
    if (i < n) out[i] = (bf16_t)in[i];
}

// ---------------- LayerNorm -> bf16 ----------------
__global__ void ln_kernel(const float* __restrict__ x, const float* __restrict__ w,
                          const float* __restrict__ b, bf16_t* __restrict__ xn) {
    const int row = blockIdx.x;          // token row, 0..4095
    const int t   = threadIdx.x;         // 256 threads
    size_t base = (size_t)row * DIM;
    float v0 = x[base + t];
    float v1 = x[base + t + 256];
    float s  = v0 + v1;
    float sq = v0*v0 + v1*v1;
    for (int off = 16; off; off >>= 1) {
        s  += __shfl_xor(s,  off, 32);
        sq += __shfl_xor(sq, off, 32);
    }
    __shared__ float ps[8], pq[8];
    int wv = t >> 5;
    if ((t & 31) == 0) { ps[wv] = s; pq[wv] = sq; }
    __syncthreads();
    s = 0.f; sq = 0.f;
    #pragma unroll
    for (int i = 0; i < 8; ++i) { s += ps[i]; sq += pq[i]; }
    float mu  = s * (1.0f / DIM);
    float var = sq * (1.0f / DIM) - mu * mu;
    float rs  = rsqrtf(var + 1e-5f);
    xn[base + t]       = (bf16_t)((v0 - mu) * rs * w[t]       + b[t]);
    xn[base + t + 256] = (bf16_t)((v1 - mu) * rs * w[t + 256] + b[t + 256]);
}

// ---------------- WMMA GEMM: C[M,N] = A[M,K] * W[N,K]^T (+bias) (+softplus) ----
// A: bf16 row-major (lda=K), W: bf16 row-major NxK, C: f32 row-major (ldc=N).
// One wave computes a 32x32 tile (2x2 WMMA fragments, operands reused 2x each).
// 128-thread block = 4 waves = 64x64 supertile.
template<int ACT>
__global__ void gemm_bf16_kernel(const bf16_t* __restrict__ A, const bf16_t* __restrict__ W,
                                 const float* __restrict__ bias, float* __restrict__ C,
                                 int M, int N, int K) {
    const int lane = threadIdx.x & 31;
    const int wave = threadIdx.x >> 5;                  // 0..3
    const int m0 = blockIdx.x * 64 + (wave >> 1) * 32;
    const int n0 = blockIdx.y * 64 + (wave & 1) * 32;

    const int lr   = lane & 15;
    const int hk8  = (lane >> 4) * 16;     // A K-chunk base (contiguous, matches B)
    const int hk16 = (lane >> 4) * 16;     // B K-chunk base

    const bf16_t* Ap0 = A + (size_t)(m0 + lr) * K;
    const bf16_t* Ap1 = Ap0 + (size_t)16 * K;
    const bf16_t* Wp0 = W + (size_t)(n0 + lr) * K;
    const bf16_t* Wp1 = Wp0 + (size_t)16 * K;

    v8f acc00 = {}, acc01 = {}, acc10 = {}, acc11 = {};
    for (int k = 0; k < K; k += 32) {
        v8bf a0lo = *(const v8bf*)(Ap0 + k + hk8);
        v8bf a0hi = *(const v8bf*)(Ap0 + k + hk8 + 8);
        v8bf a1lo = *(const v8bf*)(Ap1 + k + hk8);
        v8bf a1hi = *(const v8bf*)(Ap1 + k + hk8 + 8);
        v16bf a0 = SHUF16(a0lo, a0hi);
        v16bf a1 = SHUF16(a1lo, a1hi);
        v16bf b0 = *(const v16bf*)(Wp0 + k + hk16);
        v16bf b1 = *(const v16bf*)(Wp1 + k + hk16);
        acc00 = WMMA_BF16(a0, b0, acc00);
        acc01 = WMMA_BF16(a0, b1, acc01);
        acc10 = WMMA_BF16(a1, b0, acc10);
        acc11 = WMMA_BF16(a1, b1, acc11);
    }

    const int mb  = (lane >> 4) * 8;
    float bv0 = bias ? bias[n0 + lr]      : 0.0f;
    float bv1 = bias ? bias[n0 + 16 + lr] : 0.0f;
    #pragma unroll
    for (int r = 0; r < 8; ++r) {
        float v00 = acc00[r] + bv0;
        float v01 = acc01[r] + bv1;
        float v10 = acc10[r] + bv0;
        float v11 = acc11[r] + bv1;
        if (ACT == 1) {  // softplus
            v00 = (v00 > 20.0f) ? v00 : log1pf(__expf(v00));
            v01 = (v01 > 20.0f) ? v01 : log1pf(__expf(v01));
            v10 = (v10 > 20.0f) ? v10 : log1pf(__expf(v10));
            v11 = (v11 > 20.0f) ? v11 : log1pf(__expf(v11));
        }
        C[(size_t)(m0 + mb + r)      * N + n0 + lr]      = v00;
        C[(size_t)(m0 + mb + r)      * N + n0 + 16 + lr] = v01;
        C[(size_t)(m0 + 16 + mb + r) * N + n0 + lr]      = v10;
        C[(size_t)(m0 + 16 + mb + r) * N + n0 + 16 + lr] = v11;
    }
}

// ---------------- depthwise causal conv (k=4) + SiLU ----------------
__global__ void conv_silu_kernel(const float* __restrict__ xz, const float* __restrict__ cw,
                                 const float* __restrict__ cb, float* __restrict__ xc,
                                 bf16_t* __restrict__ xcb) {
    size_t idx = (size_t)blockIdx.x * 256 + threadIdx.x;   // B*SEQ*D_INNER
    int d = idx & (D_INNER - 1);
    int l = (idx >> 10) & (SEQ - 1);
    int b = (int)(idx >> 20);
    float acc = cb[d];
    #pragma unroll
    for (int k = 0; k < D_CONV; ++k) {
        int ls = l - (D_CONV - 1) + k;
        if (ls >= 0)
            acc += cw[d * D_CONV + k] * xz[((size_t)b * SEQ + ls) * (2 * D_INNER) + d];
    }
    float sv = acc / (1.0f + __expf(-acc));
    xc[idx]  = sv;
    xcb[idx] = (bf16_t)sv;
}

// ---------------- delta input: dbc[:,:32] + audio dt-bias -> bf16 ----------------
__global__ void deltain_kernel(const float* __restrict__ dbc, const float* __restrict__ ae,
                               const float* __restrict__ ew, const float* __restrict__ eb,
                               bf16_t* __restrict__ out) {
    int idx = blockIdx.x * 256 + threadIdx.x;  // MTOK*DT_RANK
    int r = idx & (DT_RANK - 1);
    int t = idx >> 5;
    float inv = 1.0f / (ae[t] + 1e-4f);
    float v = dbc[(size_t)t * (DT_RANK + 2 * D_STATE) + r] + inv * ew[r] + eb[r];
    out[idx] = (bf16_t)v;
}

// ---------------- fused selective scan ----------------
// one wave32 = 2 channels x 16 states. 16-step chunks: per-step operands are
// preloaded by one lane each and broadcast via shfl, so the only loop-carried
// dependence is the h = dA*h + dBx FMA chain. Gated output written once/chunk.
__global__ void scan_kernel(const float* __restrict__ delta, const float* __restrict__ xc,
                            const float* __restrict__ dbc, const float* __restrict__ xz,
                            const float* __restrict__ A_log, const float* __restrict__ Dp,
                            bf16_t* __restrict__ ymul) {
    int gw   = (int)((blockIdx.x * (size_t)blockDim.x + threadIdx.x) >> 5);
    int lane = threadIdx.x & 31;
    int n    = lane & 15;
    int p    = gw * 2 + (lane >> 4);   // (b,d) pair, 0..4095
    int b    = p >> 10;
    int d    = p & (D_INNER - 1);

    float a      = -__expf(A_log[d * D_STATE + n]);
    float dparam = Dp[d];
    float h = 0.0f;
    size_t tbase = (size_t)b * SEQ;

    for (int l0 = 0; l0 < SEQ; l0 += 16) {
        size_t tn = tbase + l0 + n;                     // this lane's step in chunk
        float dlc = delta[tn * D_INNER + d];
        float xvc = xc[tn * D_INNER + d];
        float zvc = xz[tn * (2 * D_INNER) + D_INNER + d];
        if (l0 + 16 < SEQ) {
            __builtin_prefetch(&delta[(tn + 16) * D_INNER + d], 0, 0);
            __builtin_prefetch(&xc[(tn + 16) * D_INNER + d], 0, 0);
        }
        float o = 0.0f;
        #pragma unroll
        for (int j = 0; j < 16; ++j) {
            size_t t  = tbase + l0 + j;
            float Bv = dbc[t * (DT_RANK + 2 * D_STATE) + DT_RANK + n];
            float Cv = dbc[t * (DT_RANK + 2 * D_STATE) + DT_RANK + D_STATE + n];
            float dl = __shfl(dlc, j, 16);
            float xv = __shfl(xvc, j, 16);
            float dA = __expf(dl * a);
            h = dA * h + dl * Bv * xv;
            float y = h * Cv;
            y += __shfl_xor(y, 8, 16);
            y += __shfl_xor(y, 4, 16);
            y += __shfl_xor(y, 2, 16);
            y += __shfl_xor(y, 1, 16);
            if (j == n)   // lane n keeps step n's gated output
                o = (y + dparam * xvc) * (zvc / (1.0f + __expf(-zvc)));
        }
        ymul[tn * D_INNER + d] = (bf16_t)o;
    }
}

// ---------------- host launcher ----------------
extern "C" void kernel_launch(void* const* d_in, const int* in_sizes, int n_in,
                              void* d_out, int out_size, void* d_ws, size_t ws_size,
                              hipStream_t stream) {
    const float* x     = (const float*)d_in[0];
    const float* ae    = (const float*)d_in[1];
    const float* nw    = (const float*)d_in[2];
    const float* nb    = (const float*)d_in[3];
    const float* in_w  = (const float*)d_in[4];
    const float* in_b  = (const float*)d_in[5];
    const float* cw    = (const float*)d_in[6];
    const float* cb    = (const float*)d_in[7];
    const float* xpw   = (const float*)d_in[8];
    const float* dtw   = (const float*)d_in[9];
    const float* dtb   = (const float*)d_in[10];
    const float* e2w   = (const float*)d_in[11];
    const float* e2b   = (const float*)d_in[12];
    const float* A_log = (const float*)d_in[13];
    const float* Dp    = (const float*)d_in[14];
    const float* ow    = (const float*)d_in[15];
    const float* ob    = (const float*)d_in[16];
    float* out = (float*)d_out;

    char* ws = (char*)d_ws;
    size_t off = 0;
    auto alloc = [&](size_t bytes) { char* p = ws + off; off += (bytes + 255) & ~(size_t)255; return p; };

    bf16_t* in_w_b  = (bf16_t*)alloc((size_t)2*D_INNER*DIM*2);               // 2048x512
    bf16_t* xpw_b   = (bf16_t*)alloc((size_t)(DT_RANK+2*D_STATE)*D_INNER*2); // 64x1024
    bf16_t* dtw_b   = (bf16_t*)alloc((size_t)D_INNER*DT_RANK*2);             // 1024x32
    bf16_t* ow_b    = (bf16_t*)alloc((size_t)DIM*D_INNER*2);                 // 512x1024
    bf16_t* xn_b    = (bf16_t*)alloc((size_t)MTOK*DIM*2);
    float*  xz      = (float*) alloc((size_t)MTOK*2*D_INNER*4);
    float*  xcf     = (float*) alloc((size_t)MTOK*D_INNER*4);
    bf16_t* xcb     = (bf16_t*)alloc((size_t)MTOK*D_INNER*2);
    float*  dbc     = (float*) alloc((size_t)MTOK*(DT_RANK+2*D_STATE)*4);
    bf16_t* din_b   = (bf16_t*)alloc((size_t)MTOK*DT_RANK*2);
    float*  delta   = (float*) alloc((size_t)MTOK*D_INNER*4);
    bf16_t* ymul    = (bf16_t*)alloc((size_t)MTOK*D_INNER*2);
    (void)ws_size; (void)n_in; (void)in_sizes; (void)out_size;

    // weight casts
    f2bf_kernel<<<(2*D_INNER*DIM)/256, 256, 0, stream>>>(in_w, in_w_b, 2*D_INNER*DIM);
    f2bf_kernel<<<((DT_RANK+2*D_STATE)*D_INNER)/256, 256, 0, stream>>>(xpw, xpw_b, (DT_RANK+2*D_STATE)*D_INNER);
    f2bf_kernel<<<(D_INNER*DT_RANK)/256, 256, 0, stream>>>(dtw, dtw_b, D_INNER*DT_RANK);
    f2bf_kernel<<<(DIM*D_INNER)/256, 256, 0, stream>>>(ow, ow_b, DIM*D_INNER);

    // LayerNorm
    ln_kernel<<<MTOK, 256, 0, stream>>>(x, nw, nb, xn_b);

    // in_proj: xz = xn @ in_w^T + in_b   (M=4096, N=2048, K=512)
    gemm_bf16_kernel<0><<<dim3(MTOK/64, (2*D_INNER)/64), 128, 0, stream>>>(
        xn_b, in_w_b, in_b, xz, MTOK, 2*D_INNER, DIM);

    // depthwise causal conv + SiLU
    conv_silu_kernel<<<((size_t)MTOK*D_INNER)/256, 256, 0, stream>>>(xz, cw, cb, xcf, xcb);

    // x_proj: dbc = xc @ xproj_w^T       (M=4096, N=64, K=1024, no bias)
    gemm_bf16_kernel<0><<<dim3(MTOK/64, (DT_RANK+2*D_STATE)/64), 128, 0, stream>>>(
        xcb, xpw_b, nullptr, dbc, MTOK, DT_RANK+2*D_STATE, D_INNER);

    // delta input with acoustic dt-bias
    deltain_kernel<<<(MTOK*DT_RANK)/256, 256, 0, stream>>>(dbc, ae, e2w, e2b, din_b);

    // dt_proj + softplus: delta = softplus(din @ dt_w^T + dt_b)  (M=4096, N=1024, K=32)
    gemm_bf16_kernel<1><<<dim3(MTOK/64, D_INNER/64), 128, 0, stream>>>(
        din_b, dtw_b, dtb, delta, MTOK, D_INNER, DT_RANK);

    // fused selective scan + D-skip + silu(z) gate -> bf16
    scan_kernel<<<(BATCH*D_INNER/2*32)/256, 256, 0, stream>>>(
        delta, xcf, dbc, xz, A_log, Dp, ymul);

    // out_proj: out = ymul @ out_w^T + out_b  (M=4096, N=512, K=1024)
    gemm_bf16_kernel<0><<<dim3(MTOK/64, DIM/64), 128, 0, stream>>>(
        ymul, ow_b, ob, out, MTOK, DIM, D_INNER);
}
